// Attention_84739704750733
// MI455X (gfx1250) — matrix operand order repaired and checked by reading, not compiled, not run on hardware
//
#include <hip/hip_runtime.h>
#include <hip/hip_bf16.h>

// ---------------------------------------------------------------------------
// Attention forward, MI455X (gfx1250, wave32, WMMA 16x16x32 f16 -> f32)
// B=4, N=2048, C=1024, H=16, DH=64
// TDM (tensor_load_to_lds) used for non-transposed global->LDS tiles.
// ---------------------------------------------------------------------------

#define AB_  4
#define AN_  2048
#define AC_  1024
#define AH_  16
#define ADH_ 64
#define ASCALE_ 0.125f   // DH^-0.5 = 1/8

typedef __attribute__((ext_vector_type(16))) _Float16 v16h;
typedef __attribute__((ext_vector_type(8)))  float    v8f;
typedef __attribute__((ext_vector_type(4)))  unsigned u32x4;
typedef __attribute__((ext_vector_type(8)))  int      i32x8;

// ---------------------------------------------------------------------------
// fp32 -> fp16 conversion (grid-stride)
// ---------------------------------------------------------------------------
__global__ void cvt_f32_to_f16(const float* __restrict__ src,
                               _Float16* __restrict__ dst, int n) {
    int i = blockIdx.x * blockDim.x + threadIdx.x;
    int stride = gridDim.x * blockDim.x;
    for (; i < n; i += stride) dst[i] = (_Float16)src[i];
}

// ---------------------------------------------------------------------------
// Tensor Data Mover: load a [rows x 64] f16 tile (row stride strideElems) from
// global memory into LDS with a padded 72-half (144B) LDS row stride.
// 64 halves/row = 128B = 32 DWORDs; pad 4 DWORDs after every 32 DWORDs:
//   pad_interval code 4 (=32 DWORDs), pad_amount code 3 (=4 DWORDs).
// Issue from ONE wave (EXEC is ignored; the op is per-wave). Descriptor
// Group0/Group1 per CDNA5 ISA 8.3/8.4; VADDR2/VADDR3 omitted (NULL, <=2D).
// ---------------------------------------------------------------------------
__device__ __forceinline__ void tdm_load_tile_f16(unsigned ldsAddr,
                                                  const _Float16* gsrc,
                                                  int rows, int strideElems) {
    unsigned long long ga = (unsigned long long)(size_t)gsrc;
    u32x4 g0;
    g0.x = 1u;                                  // count=1 (valid user D#)
    g0.y = ldsAddr;                             // lds_addr (bytes)
    g0.z = (unsigned)ga;                        // global_addr[31:0]
    g0.w = (unsigned)(ga >> 32) | 0x80000000u;  // global_addr[56:32] | type=2

    i32x8 g1;
    g1[0] = (1 << 16)        // data_size = 1 (2-byte elements)
          | (1 << 20)        // pad_enable
          | (4 << 22)        // pad_interval: 32 DWORDs
          | (3 << 25);       // pad_amount:   4 DWORDs
    g1[1] = (64 << 16);      // tensor_dim0 = 64 (bits[79:48], low half)
    g1[2] = (rows << 16);    // tensor_dim0 hi=0 | tensor_dim1 low16 = rows
    g1[3] = (64 << 16);      // tensor_dim1 hi=0 | tile_dim0 = 64
    g1[4] = rows & 0xFFFF;   // tile_dim1 = rows | tile_dim2 = 0
    g1[5] = strideElems;     // tensor_dim0_stride low32 (elements)
    g1[6] = 0;               // stride hi16 | tensor_dim1_stride low16
    g1[7] = 0;               // tensor_dim1_stride hi32 (unused, 2D)

    asm volatile("tensor_load_to_lds %0, %1" :: "s"(g0), "s"(g1) : "memory");
}

__device__ __forceinline__ unsigned lds_addr_of(const void* p) {
    // Generic shared pointer = {SHARED_BASE[63:32], lds_offset[31:0]}.
    return (unsigned)(size_t)p;
}

// ---------------------------------------------------------------------------
// Fragment loaders (wave32 WMMA layouts, CDNA5 ISA 7.12.2)
// ---------------------------------------------------------------------------
// A-matrix 16x32 f16: lanes 0-15 & 16-31 both hold rows M=0..15;
// this code puts pair i at K = g*16 + i*2, g = lane/16 (B uses the same k order).
__device__ __forceinline__ v16h load_a_frag(const _Float16* rowPtr, int g) {
    union { v16h v; unsigned u[8]; } f;
#pragma unroll
    for (int i = 0; i < 8; ++i) {
        int k = (g << 4) + (i << 1);
        f.u[i] = *(const unsigned*)(rowPtr + k);
    }
    return f.v;
}

// B-matrix 32x16 f16 from a [col][k]-transposed LDS tile: lane holds column
// N = lane%16; halves j = K = g*16 + j (this code's k order, as in A) -> 16 contiguous halves.
__device__ __forceinline__ v16h load_b_frag(const _Float16* p) {
    union { v16h v; uint4 q[2]; } f;
    f.q[0] = *(const uint4*)(p);
    f.q[1] = *(const uint4*)(p + 8);
    return f.v;
}

// ---------------------------------------------------------------------------
// 128x64-tile WMMA GEMM: C[M,Ncols] = A[M,K] @ Bm[K,Ncols] (f16 in, f32 acc)
// 256 threads = 8 waves; wave w owns row tile w (16 rows) x all 4 col tiles.
// K-step 64 -> 8 WMMAs per wave per barrier pair. A tile via TDM.
// mode 1: scatter to Q head layout [b,h,n,d] (f16)
// mode 2: cols<1024 -> K head layout, cols>=1024 -> V head layout (f16)
// mode 3: f32 row-major + bias (final projection)
// ---------------------------------------------------------------------------
__global__ void __launch_bounds__(256)
gemm_wmma(const _Float16* __restrict__ A, const _Float16* __restrict__ Bm,
          int K, int Ncols, int mode,
          _Float16* __restrict__ out16a, _Float16* __restrict__ out16b,
          float* __restrict__ out32, const float* __restrict__ bias) {
    __shared__ _Float16 sA [128][72];  // [row][k]  64 + 8 pad (TDM pad)
    __shared__ _Float16 sBt[64][72];   // [col][k]  transposed B tile

    const int tid  = threadIdx.x;
    const int wave = tid >> 5;
    const int lane = tid & 31;
    const int g    = lane >> 4;
    const int ln   = lane & 15;

    const int row0 = blockIdx.x * 128;
    const int col0 = blockIdx.y * 64;

    v8f acc[4] = {};

    for (int k0 = 0; k0 < K; k0 += 64) {
        __syncthreads();
        // -- A tile 128x64 via Tensor Data Mover (wave 0 issues) ----------
        if (wave == 0) {
            tdm_load_tile_f16(lds_addr_of(&sA[0][0]),
                              A + (size_t)row0 * K + k0, 128, K);
        }
        // -- B tile 64x64, manual transpose into LDS ----------------------
        {
            int kr = tid >> 2, s = (tid & 3) << 4;
            union { uint4 q[2]; _Float16 e[16]; } t;
            const uint4* gp =
                (const uint4*)(Bm + (size_t)(k0 + kr) * Ncols + col0 + s);
            t.q[0] = gp[0]; t.q[1] = gp[1];
#pragma unroll
            for (int j = 0; j < 16; ++j) sBt[s + j][kr] = t.e[j];
        }
        if (wave == 0) __builtin_amdgcn_s_wait_tensorcnt(0);
        __syncthreads();

#pragma unroll
        for (int kc = 0; kc < 64; kc += 32) {
            v16h af = load_a_frag(&sA[wave * 16 + ln][kc], g);
#pragma unroll
            for (int t = 0; t < 4; ++t) {
                v16h bf = load_b_frag(&sBt[t * 16 + ln][kc + g * 16]);
                acc[t] = __builtin_amdgcn_wmma_f32_16x16x32_f16(
                    false, af, false, bf, (short)0, acc[t], false, false);
            }
        }
    }

    // -- epilogue ---------------------------------------------------------
#pragma unroll
    for (int t = 0; t < 4; ++t) {
#pragma unroll
        for (int v = 0; v < 8; ++v) {
            int r = row0 + wave * 16 + v + 8 * g;
            int c = col0 + t * 16 + ln;
            float val = acc[t][v];
            if (mode == 3) {
                out32[(size_t)r * Ncols + c] = val + bias[c];
            } else {
                int cc = c;
                _Float16* dst = out16a;
                if (mode == 2 && cc >= AC_) { cc -= AC_; dst = out16b; }
                int b = r >> 11, n = r & (AN_ - 1);
                int h = cc >> 6, d = cc & (ADH_ - 1);
                dst[(((size_t)(b * AH_ + h) * AN_) + n) * ADH_ + d] =
                    (_Float16)val;
            }
        }
    }
}

// ---------------------------------------------------------------------------
// Flash attention: one block per (b,h, 64-query-row block).
// Q tile resident in LDS (TDM); stream 64-key K (TDM) / V (manual transpose)
// blocks; online softmax (f32) in LDS; P@V accumulated in WMMA f32 frags.
// ---------------------------------------------------------------------------
__global__ void __launch_bounds__(256)
flash_attn(const _Float16* __restrict__ Qh, const _Float16* __restrict__ Kh,
           const _Float16* __restrict__ Vh, _Float16* __restrict__ Oh) {
    __shared__ _Float16 sQ [64][72];   // [qrow][d]
    __shared__ _Float16 sK [64][72];   // [key][d]   (acts as K^T B-tile)
    __shared__ _Float16 sVt[64][72];   // [d][key]   transposed V
    __shared__ _Float16 sP [64][72];   // [qrow][key] probabilities (f16)
    __shared__ float    sS [64][68];   // [qrow][key] scores (f32)
    __shared__ float    sM[64], sL[64], sAl[64];

    const int tid  = threadIdx.x;
    const int wave = tid >> 5;
    const int lane = tid & 31;
    const int g    = lane >> 4;
    const int ln   = lane & 15;

    const int nQB = AN_ / 64;                 // 32
    const int qb  = blockIdx.x % nQB;
    const int bh  = blockIdx.x / nQB;         // b*H + h

    const _Float16* Qbase = Qh + ((size_t)bh * AN_ + qb * 64) * ADH_;
    const _Float16* Kbase = Kh + (size_t)bh * AN_ * ADH_;
    const _Float16* Vbase = Vh + (size_t)bh * AN_ * ADH_;

    // -- resident Q tile (64x64 f16) via TDM ------------------------------
    if (wave == 0) {
        tdm_load_tile_f16(lds_addr_of(&sQ[0][0]), Qbase, 64, ADH_);
        __builtin_amdgcn_s_wait_tensorcnt(0);
    }
    if (tid < 64) { sM[tid] = -1e30f; sL[tid] = 0.0f; }

    const int rt  = wave >> 1;
    const int ctb = (wave & 1) * 2;
    v8f oacc[2] = {};

    for (int kb = 0; kb < AN_ / 64; ++kb) {
        __syncthreads();
        // -- K tile [key][d] via TDM --------------------------------------
        if (wave == 0) {
            tdm_load_tile_f16(lds_addr_of(&sK[0][0]),
                              Kbase + (size_t)kb * 64 * ADH_, 64, ADH_);
        }
        // -- V tile transposed [d][key], manual ---------------------------
        {
            int key = tid >> 2, s = (tid & 3) << 4;
            union { uint4 q[2]; _Float16 e[16]; } t;
            const uint4* gp =
                (const uint4*)(Vbase + ((size_t)(kb * 64 + key)) * ADH_ + s);
            t.q[0] = gp[0]; t.q[1] = gp[1];
#pragma unroll
            for (int j = 0; j < 16; ++j) sVt[s + j][key] = t.e[j];
        }
        if (wave == 0) __builtin_amdgcn_s_wait_tensorcnt(0);
        __syncthreads();

        // -- S = Q @ K^T (64x64), each wave 2 tiles ----------------------
        v8f sacc[2] = {};
#pragma unroll
        for (int kc = 0; kc < 64; kc += 32) {
            v16h af = load_a_frag(&sQ[rt * 16 + ln][kc], g);
#pragma unroll
            for (int t = 0; t < 2; ++t) {
                v16h bf = load_b_frag(&sK[(ctb + t) * 16 + ln][kc + g * 16]);
                sacc[t] = __builtin_amdgcn_wmma_f32_16x16x32_f16(
                    false, af, false, bf, (short)0, sacc[t], false, false);
            }
        }
#pragma unroll
        for (int t = 0; t < 2; ++t)
#pragma unroll
            for (int v = 0; v < 8; ++v)
                sS[rt * 16 + v + 8 * g][(ctb + t) * 16 + ln] =
                    sacc[t][v] * ASCALE_;
        __syncthreads();

        // -- online softmax: one thread per query row ---------------------
        if (tid < 64) {
            int r = tid;
            float mold = sM[r];
            float mnew = mold;
#pragma unroll 4
            for (int j = 0; j < 64; ++j) mnew = fmaxf(mnew, sS[r][j]);
            float alpha = __expf(mold - mnew);
            float psum = 0.0f;
#pragma unroll 4
            for (int j = 0; j < 64; ++j) {
                float p = __expf(sS[r][j] - mnew);
                sP[r][j] = (_Float16)p;
                psum += p;
            }
            sM[r]  = mnew;
            sL[r]  = sL[r] * alpha + psum;
            sAl[r] = alpha;
        }
        __syncthreads();

        // -- O = O*alpha + P @ V -----------------------------------------
#pragma unroll
        for (int t = 0; t < 2; ++t)
#pragma unroll
            for (int v = 0; v < 8; ++v)
                oacc[t][v] *= sAl[rt * 16 + v + 8 * g];
#pragma unroll
        for (int kc = 0; kc < 64; kc += 32) {
            v16h af = load_a_frag(&sP[rt * 16 + ln][kc], g);
#pragma unroll
            for (int t = 0; t < 2; ++t) {
                v16h bf = load_b_frag(&sVt[(ctb + t) * 16 + ln][kc + g * 16]);
                oacc[t] = __builtin_amdgcn_wmma_f32_16x16x32_f16(
                    false, af, false, bf, (short)0, oacc[t], false, false);
            }
        }
    }
    __syncthreads();

    // -- normalize and store O in [b, n, h*DH+d] f16 layout ----------------
    const int b = bh >> 4, h = bh & (AH_ - 1);
#pragma unroll
    for (int t = 0; t < 2; ++t)
#pragma unroll
        for (int v = 0; v < 8; ++v) {
            int r = rt * 16 + v + 8 * g;
            int d = (ctb + t) * 16 + ln;
            float val = oacc[t][v] / sL[r];
            Oh[(((size_t)b * AN_) + qb * 64 + r) * AC_ + h * ADH_ + d] =
                (_Float16)val;
        }
}

// ---------------------------------------------------------------------------
// Host launcher
// ---------------------------------------------------------------------------
extern "C" void kernel_launch(void* const* d_in, const int* in_sizes, int n_in,
                              void* d_out, int out_size, void* d_ws,
                              size_t ws_size, hipStream_t stream) {
    (void)in_sizes; (void)n_in; (void)out_size; (void)ws_size;
    const float* x    = (const float*)d_in[0];
    const float* Wq   = (const float*)d_in[1];
    const float* Wkv  = (const float*)d_in[2];
    const float* Wout = (const float*)d_in[3];
    const float* bout = (const float*)d_in[4];
    float* out = (float*)d_out;

    const size_t BNC = (size_t)AB_ * AN_ * AC_;   // 8,388,608
    const size_t CC  = (size_t)AC_ * AC_;         // 1,048,576

    _Float16* ws    = (_Float16*)d_ws;
    size_t off = 0;
    _Float16* xh    = ws + off; off += BNC;
    _Float16* Wqh   = ws + off; off += CC;
    _Float16* Wkvh  = ws + off; off += 2 * CC;
    _Float16* Wouth = ws + off; off += CC;
    _Float16* Qh    = ws + off; off += BNC;
    _Float16* Kh    = ws + off; off += BNC;
    _Float16* Vh    = ws + off; off += BNC;
    _Float16* Oh    = ws + off; off += BNC;

    // 1) fp32 -> fp16 conversions
    cvt_f32_to_f16<<<4096, 256, 0, stream>>>(x,    xh,    (int)BNC);
    cvt_f32_to_f16<<<1024, 256, 0, stream>>>(Wq,   Wqh,   (int)CC);
    cvt_f32_to_f16<<<2048, 256, 0, stream>>>(Wkv,  Wkvh,  (int)(2 * CC));
    cvt_f32_to_f16<<<1024, 256, 0, stream>>>(Wout, Wouth, (int)CC);

    const int M = AB_ * AN_;  // 8192 rows

    // 2) Q projection -> head layout
    gemm_wmma<<<dim3(M / 128, AC_ / 64), 256, 0, stream>>>(
        xh, Wqh, AC_, AC_, 1, Qh, nullptr, nullptr, nullptr);
    // 3) fused KV projection -> head layouts
    gemm_wmma<<<dim3(M / 128, (2 * AC_) / 64), 256, 0, stream>>>(
        xh, Wkvh, AC_, 2 * AC_, 2, Kh, Vh, nullptr, nullptr);

    // 4) flash attention: one block per (b,h,64-query block)
    flash_attn<<<AB_ * AH_ * (AN_ / 64), 256, 0, stream>>>(Qh, Kh, Vh, Oh);

    // 5) output projection + bias -> f32 out
    gemm_wmma<<<dim3(M / 128, AC_ / 64), 256, 0, stream>>>(
        Oh, Wouth, AC_, AC_, 3, nullptr, nullptr, out, bout);
}
